// Self_Attention_83425444757851
// MI455X (gfx1250) — matrix operand order repaired and checked by reading, not compiled, not run on hardware
//
#include <hip/hip_runtime.h>
#include <hip/hip_bf16.h>

// ---------------------------------------------------------------------------
// Types for CDNA5 WMMA (wave32): 16x16x32 f16 -> f32, and TDM descriptors
// ---------------------------------------------------------------------------
typedef __attribute__((ext_vector_type(16))) _Float16 v16h;
typedef __attribute__((ext_vector_type(4)))  _Float16 v4h;
typedef __attribute__((ext_vector_type(8)))  float    v8f;
typedef __attribute__((ext_vector_type(4)))  unsigned int u32x4;
typedef __attribute__((ext_vector_type(8)))  int      i32x8;
typedef __attribute__((ext_vector_type(4)))  int      i32x4;

#define WMMA_F16(a, b, c) \
  __builtin_amdgcn_wmma_f32_16x16x32_f16(false, (a), false, (b), (short)0, (c), false, false)

// A fragment (16x32 f16, row-major source, stride `ld` halves).
// ISA layout: lane<16 -> row=lane, K {0..7, 16..23}; lane>=16 -> row=lane-16, K {8..15, 24..31}
__device__ __forceinline__ v16h load_a_frag(const _Float16* base, int ld) {
  const int lane = threadIdx.x & 31;
  const int r = lane & 15, hi = lane >> 4;
  const _Float16* p = base + r * ld + hi * 8;
  union { v16h v; u32x4 q[2]; } u;
  u.q[0] = *(const u32x4*)(p);        // K: hi*8 .. +7
  u.q[1] = *(const u32x4*)(p + 16);   // K: 16 + hi*8 .. +7
  return u.v;
}

// B fragment (32x16 f16) from a TRANSPOSED weight (N x K row-major, stride `ld` halves).
// ISA layout: lane<16 -> col=lane, K {0..7, 16..23}; lane>=16 -> col=lane-16, K {8..15, 24..31}
__device__ __forceinline__ v16h load_b_frag(const _Float16* base, int ld) {
  const int lane = threadIdx.x & 31;
  const int n = lane & 15, hi = lane >> 4;
  const _Float16* p = base + n * ld + hi * 8;
  union { v16h v; u32x4 q[2]; } u;
  u.q[0] = *(const u32x4*)(p);
  u.q[1] = *(const u32x4*)(p + 16);
  return u.v;
}

// ---------------------------------------------------------------------------
// Tensor Data Mover: 2D tile (tile_d0 x tile_d1 elements, 2-byte data) from
// global (row stride `stride0` elements) into LDS at byte offset `lds_addr`.
// Descriptor packing per CDNA5 ISA ch.10.8 (D# groups 0/1; groups 2/3 unused
// for 2D tiles -> zero). One instruction per wave; tracked by TENSORcnt.
// This toolchain exposes the 6-arg builtin form (extra int32x8 group, zeroed).
// ---------------------------------------------------------------------------
__device__ __forceinline__ void tdm_load_2d_f16(unsigned lds_addr, const void* gptr,
                                                unsigned tile_d0, unsigned tile_d1,
                                                unsigned tensor_d0, unsigned tensor_d1,
                                                unsigned long long stride0) {
  unsigned long long ga = (unsigned long long)gptr;
  u32x4 g0;
  g0[0] = 1u;                                            // count=1 valid user D#
  g0[1] = lds_addr;                                      // LDS byte address
  g0[2] = (unsigned)(ga & 0xFFFFFFFFu);                  // global_addr[31:0]
  g0[3] = (unsigned)((ga >> 32) & 0x01FFFFFFu) | 0x80000000u;  // addr[56:32] | type=2
  i32x8 g1;
  g1[0] = 0x00010000;                                    // wg_mask=0, data_size=1 (2B)
  g1[1] = (int)((tensor_d0 & 0xFFFFu) << 16);            // tensor_dim0[15:0]
  g1[2] = (int)((tensor_d0 >> 16) | ((tensor_d1 & 0xFFFFu) << 16));
  g1[3] = (int)((tensor_d1 >> 16) | (tile_d0 << 16));    // tile_dim0
  g1[4] = (int)(tile_d1);                                // tile_dim1 (tile_dim2=0 -> 2D)
  g1[5] = (int)(unsigned)(stride0 & 0xFFFFFFFFu);        // tensor_dim0_stride[31:0]
  g1[6] = (int)(unsigned)((stride0 >> 32) & 0xFFFFu);    // stride[47:32]; dim1_stride=0
  g1[7] = 0;
  i32x4 z4 = {0, 0, 0, 0};
  i32x8 z8 = {0, 0, 0, 0, 0, 0, 0, 0};
  __builtin_amdgcn_tensor_load_to_lds(g0, g1, z4, z4, z8, 0);
}

// ---------------------------------------------------------------------------
// Kernel 0: f32 (rows x cols) -> f16 transposed (cols x rows)
// ---------------------------------------------------------------------------
__global__ void transpose_cvt(const float* __restrict__ src, _Float16* __restrict__ dst,
                              int rows, int cols) {
  long idx = (long)blockIdx.x * blockDim.x + threadIdx.x;
  long total = (long)rows * cols;
  if (idx >= total) return;
  long c = idx / rows;      // dst row (output feature)
  long r = idx % rows;      // dst col (k)
  dst[c * rows + r] = (_Float16)src[r * cols + c];
}

// ---------------------------------------------------------------------------
// Kernel 1: per-block = 8 batches (48 rows). 96 threads = 3 waves; wave w owns
// the 16-row M-tile [16w, 16w+16). Fuses qkv GEMM, attention, LN1, W_fp GEMM,
// LN2, and emits f16 rows of f (32768 x 3072) for the final GEMM.
// ---------------------------------------------------------------------------
#define GB 8  // batches per block

__global__ __launch_bounds__(96) void stage1(
    const float* __restrict__ x, const float* __restrict__ b_qkv,
    const float* __restrict__ g1, const float* __restrict__ be1,
    const float* __restrict__ b_fp, const float* __restrict__ g2,
    const float* __restrict__ be2, const _Float16* __restrict__ WqkvT,
    const _Float16* __restrict__ WfpT, _Float16* __restrict__ F) {
  extern __shared__ char smem[];
  _Float16* xf16 = (_Float16*)(smem);                       // 48*128 f16 = 12288 B
  float*    qkvS = (float*)(smem + 12288);                  // 48*384 f32 = 73728 B (reused as n2)
  float*    mS   = (float*)(smem + 12288 + 73728);          // 48*256 f32 = 49152 B
  _Float16* nln  = (_Float16*)(smem + 12288 + 73728 + 49152); // 48*256 f16 = 24576 B
  float*    attnS= (float*)(smem + 12288 + 73728 + 49152 + 24576); // 288 f32

  const int t = threadIdx.x;
  const int wave = t >> 5, lane = t & 31;
  const long b0 = (long)blockIdx.x * GB;
  const float* xblk = x + b0 * 6 * 128;   // 48 contiguous rows of x

  // ---- x -> f16 in LDS (A matrix of qkv GEMM) ----
  for (int i = t; i < 48 * 128 / 4; i += 96) {
    float4 v = ((const float4*)xblk)[i];
    v4h hv = {(_Float16)v.x, (_Float16)v.y, (_Float16)v.z, (_Float16)v.w};
    ((v4h*)xf16)[i] = hv;
  }
  __syncthreads();

  // ---- qkv = x @ W_qkv + b_qkv : 48x384, K=128 (WMMA) ----
  {
    const _Float16* Abase = xf16 + wave * 16 * 128;
    const int hi = lane >> 4;
    for (int nt = 0; nt < 24; ++nt) {
      v8f acc = {};
      for (int kt = 0; kt < 4; ++kt) {
        v16h a = load_a_frag(Abase + kt * 32, 128);
        v16h b = load_b_frag(WqkvT + (nt * 16) * 128 + kt * 32, 128);
        acc = WMMA_F16(a, b, acc);
      }
      const int col = nt * 16 + (lane & 15);
      const float bias = b_qkv[col];
      for (int v = 0; v < 8; ++v) {
        int row = wave * 16 + hi * 8 + v;
        qkvS[row * 384 + col] = acc[v] + bias;
      }
    }
  }
  __syncthreads();

  // ---- attention scores: attn[b] = Q_b @ K_b^T  (8 * 6x6, K=128, VALU f32) ----
  for (int idx = t; idx < GB * 36; idx += 96) {
    int b = idx / 36, rem = idx % 36;
    int qi = rem / 6, ki = rem % 6;
    const float* Q = qkvS + (b * 6 + qi) * 384;
    const float* K = qkvS + (b * 6 + ki) * 384 + 128;
    float s = 0.f;
    #pragma unroll 8
    for (int d = 0; d < 128; ++d) s += Q[d] * K[d];
    attnS[idx] = s;
  }
  __syncthreads();

  // ---- m = concat(x, attn @ V) : 48x256 ----
  {
    int r = t >> 1;                 // row 0..47
    int b = r / 6, qi = r % 6;
    int d0 = (t & 1) * 64;
    const float* arow = attnS + b * 36 + qi * 6;
    for (int d = d0; d < d0 + 64; ++d) {
      float s = 0.f;
      #pragma unroll
      for (int k = 0; k < 6; ++k) s += arow[k] * qkvS[(b * 6 + k) * 384 + 256 + d];
      mS[r * 256 + 128 + d] = s;
      mS[r * 256 + d] = xblk[r * 128 + d];
    }
  }
  __syncthreads();

  // ---- LN1 over 256 -> nln (f16 A for the W_fp GEMM) ----
  if (t < 48) {
    const float* row = mS + t * 256;
    float mu = 0.f;
    for (int c = 0; c < 256; ++c) mu += row[c];
    mu *= (1.f / 256.f);
    float var = 0.f;
    for (int c = 0; c < 256; ++c) { float d = row[c] - mu; var += d * d; }
    var *= (1.f / 256.f);
    float rs = rsqrtf(var + 1e-5f);
    for (int c = 0; c < 256; ++c)
      nln[t * 256 + c] = (_Float16)((row[c] - mu) * rs * g1[c] + be1[c]);
  }
  __syncthreads();

  // ---- n2 = LN1(m) @ W_fp + b_fp : 48x256, K=256 (WMMA); reuse qkvS storage ----
  float* n2 = qkvS;
  {
    const _Float16* Abase = nln + wave * 16 * 256;
    const int hi = lane >> 4;
    for (int nt = 0; nt < 16; ++nt) {
      v8f acc = {};
      for (int kt = 0; kt < 8; ++kt) {
        v16h a = load_a_frag(Abase + kt * 32, 256);
        v16h b = load_b_frag(WfpT + (nt * 16) * 256 + kt * 32, 256);
        acc = WMMA_F16(a, b, acc);
      }
      const int col = nt * 16 + (lane & 15);
      const float bias = b_fp[col];
      for (int v = 0; v < 8; ++v) {
        int row = wave * 16 + hi * 8 + v;
        n2[row * 256 + col] = acc[v] + bias;
      }
    }
  }
  __syncthreads();

  // ---- LN2 over concat(n2, m) (512) and emit f16 row of F ----
  if (t < 48) {
    const float* na = n2 + t * 256;
    const float* mb = mS + t * 256;
    float mu = 0.f;
    for (int c = 0; c < 256; ++c) mu += na[c] + mb[c];
    mu *= (1.f / 512.f);
    float var = 0.f;
    for (int c = 0; c < 256; ++c) {
      float d1 = na[c] - mu, d2 = mb[c] - mu;
      var += d1 * d1 + d2 * d2;
    }
    var *= (1.f / 512.f);
    float rs = rsqrtf(var + 1e-5f);
    int b = t / 6, s = t % 6;
    _Float16* outp = F + (b0 + b) * 3072 + s * 512;
    for (int c = 0; c < 256; ++c)
      outp[c] = (_Float16)((na[c] - mu) * rs * g2[c] + be2[c]);
    for (int c = 0; c < 256; ++c)
      outp[256 + c] = (_Float16)((mb[c] - mu) * rs * g2[256 + c] + be2[256 + c]);
  }
}

// ---------------------------------------------------------------------------
// Kernel 2: out = relu(F(32768x3072) @ W_s + b_s), f16 WMMA, f32 epilogue.
// Block: 256 thr (8 waves), tile 128(M) x 128(N), K in 64-wide slabs that are
// DMA'd global->LDS by the Tensor Data Mover (wave0: A, wave1: B), double
// buffered, one workgroup barrier per slab. Wave (wm,wn) owns 32x64 = 2x4 accs.
// ---------------------------------------------------------------------------
#define KSTEP 64

__global__ __launch_bounds__(256) void gemm_out(
    const _Float16* __restrict__ F, const _Float16* __restrict__ WsT,
    const float* __restrict__ b_s, float* __restrict__ out) {
  __shared__ _Float16 As[2][128 * KSTEP];   // 2 x 16 KB
  __shared__ _Float16 Bs[2][128 * KSTEP];   // 2 x 16 KB

  const int t = threadIdx.x, lane = t & 31, wave = t >> 5;
  const int wm = wave >> 1, wn = wave & 1;
  const long m0 = (long)blockIdx.y * 128;
  const int  n0 = blockIdx.x * 128;

  const unsigned ldsA0 = (unsigned)(unsigned long long)&As[0][0];
  const unsigned ldsA1 = (unsigned)(unsigned long long)&As[1][0];
  const unsigned ldsB0 = (unsigned)(unsigned long long)&Bs[0][0];
  const unsigned ldsB1 = (unsigned)(unsigned long long)&Bs[1][0];

  v8f acc[2][4] = {};

  // prologue: DMA slab 0 into buffer 0
  if (wave == 0)
    tdm_load_2d_f16(ldsA0, F + m0 * 3072, KSTEP, 128, 3072, 128, 3072);
  else if (wave == 1)
    tdm_load_2d_f16(ldsB0, WsT + (long)n0 * 3072, KSTEP, 128, 3072, 128, 3072);
  if (wave < 2) __builtin_amdgcn_s_wait_tensorcnt(0);
  __syncthreads();

  for (int k0 = 0; k0 < 3072; k0 += KSTEP) {
    const int cur = (k0 / KSTEP) & 1;
    const int nxt = cur ^ 1;
    const bool more = (k0 + KSTEP) < 3072;

    // DMA next slab into the other buffer while we compute on this one
    if (more) {
      if (wave == 0)
        tdm_load_2d_f16(nxt ? ldsA1 : ldsA0, F + m0 * 3072 + k0 + KSTEP,
                        KSTEP, 128, 3072, 128, 3072);
      else if (wave == 1)
        tdm_load_2d_f16(nxt ? ldsB1 : ldsB0, WsT + (long)n0 * 3072 + k0 + KSTEP,
                        KSTEP, 128, 3072, 128, 3072);
    }

    // compute: 2 K-subslabs of 32 -> 16 WMMAs per wave per iteration
    #pragma unroll
    for (int kk = 0; kk < KSTEP; kk += 32) {
      v16h a[2], b[4];
      #pragma unroll
      for (int i = 0; i < 2; ++i)
        a[i] = load_a_frag(&As[cur][(wm * 32 + i * 16) * KSTEP + kk], KSTEP);
      #pragma unroll
      for (int j = 0; j < 4; ++j)
        b[j] = load_b_frag(&Bs[cur][(wn * 64 + j * 16) * KSTEP + kk], KSTEP);
      #pragma unroll
      for (int i = 0; i < 2; ++i)
        #pragma unroll
        for (int j = 0; j < 4; ++j)
          acc[i][j] = WMMA_F16(a[i], b[j], acc[i][j]);
    }

    if (more && wave < 2) __builtin_amdgcn_s_wait_tensorcnt(0);
    __syncthreads();
  }

  // epilogue: bias + relu, f32 store
  const int hi = lane >> 4, nl = lane & 15;
  #pragma unroll
  for (int i = 0; i < 2; ++i)
    #pragma unroll
    for (int j = 0; j < 4; ++j) {
      const int col = n0 + wn * 64 + j * 16 + nl;
      const float bias = b_s[col];
      #pragma unroll
      for (int v = 0; v < 8; ++v) {
        long row = m0 + wm * 32 + i * 16 + hi * 8 + v;
        float val = acc[i][j][v] + bias;
        out[row * 1024 + col] = val > 0.f ? val : 0.f;
      }
    }
}

// ---------------------------------------------------------------------------
// Host-side launcher
// ---------------------------------------------------------------------------
extern "C" void kernel_launch(void* const* d_in, const int* in_sizes, int n_in,
                              void* d_out, int out_size, void* d_ws, size_t ws_size,
                              hipStream_t stream) {
  const float* x     = (const float*)d_in[0];
  const float* W_qkv = (const float*)d_in[1];
  const float* b_qkv = (const float*)d_in[2];
  const float* g1    = (const float*)d_in[3];
  const float* be1   = (const float*)d_in[4];
  const float* W_fp  = (const float*)d_in[5];
  const float* b_fp  = (const float*)d_in[6];
  const float* g2    = (const float*)d_in[7];
  const float* be2   = (const float*)d_in[8];
  const float* W_s   = (const float*)d_in[9];
  const float* b_s   = (const float*)d_in[10];
  float* out = (float*)d_out;

  // workspace layout (f16):
  //   F      : 32768 x 3072           = 201326592 B
  //   WsT    : 1024  x 3072           =   6291456 B
  //   WqkvT  : 384   x 128            =     98304 B
  //   WfpT   : 256   x 256            =    131072 B
  char* ws = (char*)d_ws;
  _Float16* F     = (_Float16*)(ws);
  _Float16* WsT   = (_Float16*)(ws + 201326592UL);
  _Float16* WqkvT = (_Float16*)(ws + 207618048UL);
  _Float16* WfpT  = (_Float16*)(ws + 207716352UL);

  transpose_cvt<<<(128 * 384 + 255) / 256, 256, 0, stream>>>(W_qkv, WqkvT, 128, 384);
  transpose_cvt<<<(256 * 256 + 255) / 256, 256, 0, stream>>>(W_fp, WfpT, 256, 256);
  transpose_cvt<<<(3072 * 1024 + 255) / 256, 256, 0, stream>>>(W_s, WsT, 3072, 1024);

  const size_t smem1 = 12288 + 73728 + 49152 + 24576 + 1152;  // 160896 B (< 320 KB/WGP)
  (void)hipFuncSetAttribute((const void*)stage1,
                            hipFuncAttributeMaxDynamicSharedMemorySize, (int)smem1);
  stage1<<<32768 / GB, 96, smem1, stream>>>(x, b_qkv, g1, be1, b_fp, g2, be2,
                                            WqkvT, WfpT, F);

  gemm_out<<<dim3(1024 / 128, 32768 / 128), 256, 0, stream>>>(F, WsT, b_s, out);
}
